// Attention_27376121544753
// MI455X (gfx1250) — compile-verified, matrix operand order checked by reading, not run on hardware
//
#include <hip/hip_runtime.h>

typedef __attribute__((ext_vector_type(16))) _Float16 v16h;
typedef __attribute__((ext_vector_type(8)))  float    v8f;

#define B_   2
#define S_   2048
#define H_   32
#define KVH_ 8
#define D_   128

__device__ __forceinline__ unsigned pack2h(_Float16 a, _Float16 b) {
    union { unsigned u; _Float16 h[2]; } x;
    x.h[0] = a; x.h[1] = b;
    return x.u;
}

// ---------------------------------------------------------------------------
// Pack K (f32 -> f16) into WMMA B-fragment layout for Q*K^T.
// Fragment id: ((b*KVH + kvh)*(S/16) + kblk)*4 + dchunk ; 512 halves each.
// Element (lane,p): key = kblk*16 + lane%16 ; d = 32*c + (p/8)*16 + (lane/16)*8 + p%8
// ---------------------------------------------------------------------------
__global__ void pack_k_kernel(const float* __restrict__ xk, _Float16* __restrict__ kws) {
    const long long e = (long long)blockIdx.x * blockDim.x + threadIdx.x; // < 4194304
    const int lanep = (int)(e & 511);
    const int fi    = (int)(e >> 9);
    const int p     = lanep & 15;
    const int lane  = lanep >> 4;
    const int c     = fi & 3;
    int rest        = fi >> 2;
    const int kblk  = rest & 127; rest >>= 7;
    const int kvh   = rest & 7;
    const int b     = rest >> 3;
    const int key   = (kblk << 4) + (lane & 15);
    const int d     = 32 * c + ((p >> 3) << 4) + ((lane >> 4) << 3) + (p & 7);
    kws[e] = (_Float16)xk[(((size_t)(b * S_ + key)) * KVH_ + kvh) * D_ + d];
}

// ---------------------------------------------------------------------------
// Pack V (f32 -> f16) into WMMA B-fragment layout for P*V.
// Fragment id: ((b*KVH + kvh)*(S/32) + j)*8 + n ; 512 halves each.
// Element (lane,p): key = j*32 + (p/8)*16 + (lane/16)*8 + p%8 ; d = n*16 + lane%16
// ---------------------------------------------------------------------------
__global__ void pack_v_kernel(const float* __restrict__ xv, _Float16* __restrict__ vws) {
    const long long e = (long long)blockIdx.x * blockDim.x + threadIdx.x; // < 4194304
    const int lanep = (int)(e & 511);
    const int fi    = (int)(e >> 9);
    const int p     = lanep & 15;
    const int lane  = lanep >> 4;
    const int n     = fi & 7;
    int rest        = fi >> 3;
    const int j     = rest & 63; rest >>= 6;
    const int kvh   = rest & 7;
    const int b     = rest >> 3;
    const int key   = (j << 5) + ((p >> 3) << 4) + ((lane >> 4) << 3) + (p & 7);
    const int d     = (n << 4) + (lane & 15);
    vws[e] = (_Float16)xv[(((size_t)(b * S_ + key)) * KVH_ + kvh) * D_ + d];
}

// ---------------------------------------------------------------------------
// kv_new: copy the paged cache, then scatter new K/V at select_index.
// ---------------------------------------------------------------------------
__global__ void copy_kv_kernel(const float* __restrict__ kvbuf, float* __restrict__ out2) {
    const size_t tid = (size_t)blockIdx.x * blockDim.x + threadIdx.x; // < 16777216
    out2[tid] = kvbuf[tid];
}

__global__ void scatter_kv_kernel(const float* __restrict__ xk, const float* __restrict__ xv,
                                  const int* __restrict__ sel, float* __restrict__ out2) {
    const size_t tid = (size_t)blockIdx.x * blockDim.x + threadIdx.x; // < B*S*KVH*D = 4194304
    const int token = (int)(tid >> 10);   // / (KVH*D)
    const int rem   = (int)(tid & 1023);
    const int slot  = sel[token];
    out2[(size_t)slot * 2048 + rem]        = xk[tid];   // K half
    out2[(size_t)slot * 2048 + 1024 + rem] = xv[tid];   // V half
}

// ---------------------------------------------------------------------------
// Flash attention: one wave (32 threads) per 16-query tile of one (b,h).
// Causal loop over 32-key blocks; QK^T and P*V via v_wmma_f32_16x16x32_f16.
// ---------------------------------------------------------------------------
__global__ __launch_bounds__(32)
void attn_kernel(const float* __restrict__ xq,
                 const _Float16* __restrict__ kws,
                 const _Float16* __restrict__ vws,
                 float* __restrict__ out) {
    __shared__ unsigned plds[256];   // 16 rows x 16 half2-pairs (P re-layout)

    const int t  = threadIdx.x;
    const int tl = t & 15;
    const int tg = t >> 4;

    const int bid = blockIdx.x;           // B*H*(S/16) = 8192
    const int i   = bid & 127;            // query tile
    const int h   = (bid >> 7) & 31;      // query head
    const int b   = bid >> 12;
    const int kvh = h >> 2;               // G = 4

    const float scale = 0.08838834764831845f; // 1/sqrt(128)

    // ---- Q fragments (A-matrix 16x32 f16), scaled at load ----
    v16h qa[4];
    {
        const int qrow = (i << 4) + tl;
        const float* qptr = xq + (((size_t)(b * S_ + qrow)) * H_ + h) * D_;
        #pragma unroll
        for (int c = 0; c < 4; ++c) {
            const float4 f0 = *(const float4*)(qptr + 32 * c + tg * 8);
            const float4 f1 = *(const float4*)(qptr + 32 * c + tg * 8 + 4);
            const float4 f2 = *(const float4*)(qptr + 32 * c + 16 + tg * 8);
            const float4 f3 = *(const float4*)(qptr + 32 * c + 16 + tg * 8 + 4);
            const float r0[8] = {f0.x, f0.y, f0.z, f0.w, f1.x, f1.y, f1.z, f1.w};
            const float r1[8] = {f2.x, f2.y, f2.z, f2.w, f3.x, f3.y, f3.z, f3.w};
            #pragma unroll
            for (int p = 0; p < 8; ++p) {
                qa[c][p]     = (_Float16)(r0[p] * scale);
                qa[c][p + 8] = (_Float16)(r1[p] * scale);
            }
        }
    }

    // ---- accumulators + online-softmax state ----
    v8f o[8];
    float m8[8], l8[8];
    #pragma unroll
    for (int n = 0; n < 8; ++n) o[n] = (v8f){0.f, 0.f, 0.f, 0.f, 0.f, 0.f, 0.f, 0.f};
    #pragma unroll
    for (int r = 0; r < 8; ++r) { m8[r] = -3.0e38f; l8[r] = 0.f; }

    const int jmax = (((i << 4) + 15) >> 5);
    const size_t kbase = (size_t)(b * KVH_ + kvh) * (S_ / 16) * 4 * 512;
    const size_t vbase = (size_t)(b * KVH_ + kvh) * (S_ / 32) * 8 * 512;

    for (int j = 0; j <= jmax; ++j) {
        // ---- scores: two 16x16 tiles (keys 32j..32j+15 and +16..+31) ----
        v8f c0 = (v8f){0.f, 0.f, 0.f, 0.f, 0.f, 0.f, 0.f, 0.f};
        v8f c1 = (v8f){0.f, 0.f, 0.f, 0.f, 0.f, 0.f, 0.f, 0.f};
        #pragma unroll
        for (int c = 0; c < 4; ++c) {
            const v16h kb0 = *(const v16h*)(kws + kbase + ((size_t)(2 * j) * 4 + c) * 512 + t * 16);
            c0 = __builtin_amdgcn_wmma_f32_16x16x32_f16(false, qa[c], false, kb0, (short)0, c0, false, false);
            const v16h kb1 = *(const v16h*)(kws + kbase + ((size_t)(2 * j + 1) * 4 + c) * 512 + t * 16);
            c1 = __builtin_amdgcn_wmma_f32_16x16x32_f16(false, qa[c], false, kb1, (short)0, c1, false, false);
        }

        // ---- causal mask (only last block can cross the diagonal) ----
        if (j == jmax) {
            #pragma unroll
            for (int r = 0; r < 8; ++r) {
                const int q = (i << 4) + tg * 8 + r;        // C-layout row
                if (32 * j + tl > q)      c0[r] = -1.0e30f;
                if (32 * j + 16 + tl > q) c1[r] = -1.0e30f;
            }
        }

        // ---- online softmax (rows live across 16 lanes within one VGPR) ----
        #pragma unroll
        for (int r = 0; r < 8; ++r) {
            float mr = fmaxf(c0[r], c1[r]);
            mr = fmaxf(mr, __shfl_xor(mr, 8, 32));
            mr = fmaxf(mr, __shfl_xor(mr, 4, 32));
            mr = fmaxf(mr, __shfl_xor(mr, 2, 32));
            mr = fmaxf(mr, __shfl_xor(mr, 1, 32));
            const float mnew = fmaxf(m8[r], mr);
            const float corr = __expf(m8[r] - mnew);
            const float p0 = __expf(c0[r] - mnew);
            const float p1 = __expf(c1[r] - mnew);
            float rs = p0 + p1;
            rs += __shfl_xor(rs, 8, 32);
            rs += __shfl_xor(rs, 4, 32);
            rs += __shfl_xor(rs, 2, 32);
            rs += __shfl_xor(rs, 1, 32);
            l8[r] = l8[r] * corr + rs;
            m8[r] = mnew;
            #pragma unroll
            for (int n = 0; n < 8; ++n) o[n][r] *= corr;
            // P re-layout: pair (P[M][k], P[M][k+16]) at pair-index M*16 + k
            plds[(tg * 8 + r) * 16 + tl] = pack2h((_Float16)p0, (_Float16)p1);
        }
        __syncthreads();

        // ---- rebuild P as A-fragment (16x32 f16) from LDS ----
        v16h pa;
        {
            const unsigned* pb = plds + tl * 16 + tg * 8;
            #pragma unroll
            for (int q = 0; q < 8; ++q) {
                union { unsigned u; _Float16 hh[2]; } x;
                x.u = pb[q];
                pa[q]     = x.hh[0];
                pa[q + 8] = x.hh[1];
            }
        }
        __syncthreads();

        // ---- P*V: 8 d-chunks of 16 columns ----
        #pragma unroll
        for (int n = 0; n < 8; ++n) {
            const v16h vb = *(const v16h*)(vws + vbase + ((size_t)j * 8 + n) * 512 + t * 16);
            o[n] = __builtin_amdgcn_wmma_f32_16x16x32_f16(false, pa, false, vb, (short)0, o[n], false, false);
        }
    }

    // ---- normalize + store: out[b, 16i+M, h*128 + 16n + tl] ----
    const size_t obase = ((size_t)b * S_ + (size_t)(i << 4)) * (H_ * D_) + (size_t)h * D_;
    #pragma unroll
    for (int n = 0; n < 8; ++n) {
        #pragma unroll
        for (int r = 0; r < 8; ++r) {
            const int M = tg * 8 + r;
            out[obase + (size_t)M * (H_ * D_) + n * 16 + tl] = o[n][r] / l8[r];
        }
    }
}

// ---------------------------------------------------------------------------
extern "C" void kernel_launch(void* const* d_in, const int* in_sizes, int n_in,
                              void* d_out, int out_size, void* d_ws, size_t ws_size,
                              hipStream_t stream) {
    const float* xq    = (const float*)d_in[0];
    const float* xk    = (const float*)d_in[1];
    const float* xv    = (const float*)d_in[2];
    const float* kvbuf = (const float*)d_in[3];
    const int*   sel   = (const int*)d_in[4];

    float* out  = (float*)d_out;
    float* out2 = out + (size_t)B_ * S_ * H_ * D_;          // kv_new region

    _Float16* kws = (_Float16*)d_ws;                         // 8 MB
    _Float16* vws = kws + (size_t)B_ * KVH_ * S_ * D_;       // 8 MB

    // f16 fragment packing of K and V (4,194,304 elements each)
    pack_k_kernel<<<16384, 256, 0, stream>>>(xk, kws);
    pack_v_kernel<<<16384, 256, 0, stream>>>(xv, vws);

    // kv_new = kv_buffer; kv_new[select_index] = (k, v)
    copy_kv_kernel<<<65536, 256, 0, stream>>>(kvbuf, out2);
    scatter_kv_kernel<<<16384, 256, 0, stream>>>(xk, xv, sel, out2);

    // causal GQA flash attention: 8192 query tiles, 1 wave each
    attn_kernel<<<B_ * H_ * (S_ / 16), 32, 0, stream>>>(xq, kws, vws, out);
}
